// OptimizedCrossAttention_41102837023287
// MI455X (gfx1250) — matrix operand order repaired and checked by reading, not compiled, not run on hardware
//
#include <hip/hip_runtime.h>

// ---------------------------------------------------------------------------
// OptimizedCrossAttention for MI455X (gfx1250, wave32, WMMA bf16 16x16x32)
//
// Pipeline:
//   1) transpose_f32_to_bf16 : query/key_value [B,C,S] -> bf16 [B,S,C]
//                              W_qkv [C,3C]    -> bf16 Wt [3C,C]
//   2) qkv_gemm_kernel       : Q=[B,NH,S,HD], K=[B,NH,S,HD], Vt=[B,NH,HD,S]
//                              (bf16, f32 accumulate, bias fused)
//   3) attn_kernel           : flash attention, online softmax, writes
//                              f32 output in [B,C,H,W] layout
// ---------------------------------------------------------------------------

#define DEV __device__ __forceinline__

typedef __attribute__((ext_vector_type(16))) __bf16 v16bf;
typedef __attribute__((ext_vector_type(8)))  __bf16 v8bf;
typedef __attribute__((ext_vector_type(8)))  float  v8f;

static constexpr int Bb = 8;
static constexpr int Cc = 512;
static constexpr int Ss = 1024;          // H*W
static constexpr int NH = 8;
static constexpr int HD = 64;            // Cc / NH
static constexpr float SCALE = 0.125f;   // HD^-0.5

// --------------------------- WMMA fragment loads ---------------------------
// A-matrix 16x32 bf16 (M x K), row-major source with row stride ld (elems).
// Per ISA 7.12.2: lanes 0-15 hold M=lane, K = {0..7, 16..23};
//                 lanes 16-31 hold M=lane-16, K = {8..15, 24..31}.
DEV v16bf load_a_frag(const __bf16* base, int ld, int m0, int k0) {
  const int lane = threadIdx.x & 31;
  const int m  = m0 + (lane & 15);
  const int kk = k0 + ((lane >> 4) << 3);          // +8 for upper half-wave
  union { v16bf v; v8bf h[2]; } u;
  u.h[0] = *(const v8bf*)(base + (long)m * ld + kk);
  u.h[1] = *(const v8bf*)(base + (long)m * ld + kk + 16);
  return u.v;
}

// B-matrix 32x16 bf16 (K x N) where B[k][n] = srcT[n][k]; srcT row-major with
// row stride ld. Lanes 0-15: N=lane, K={0..7,16..23}; lanes 16-31: N=lane-16, K={8..15,24..31}.
// Each lane reads 16 bf16 (two b128 loads).
DEV v16bf load_b_fragT(const __bf16* baseT, long ld, int k0, int n0) {
  const int lane = threadIdx.x & 31;
  const int n  = n0 + (lane & 15);
  const int kk = k0 + ((lane >> 4) << 3);          // +8 for upper half-wave
  union { v16bf v; v8bf h[2]; } u;
  u.h[0] = *(const v8bf*)(baseT + (long)n * ld + kk);
  u.h[1] = *(const v8bf*)(baseT + (long)n * ld + kk + 16);
  return u.v;
}

DEV v8f wmma_bf16(v16bf a, v16bf b, v8f c) {
  return __builtin_amdgcn_wmma_f32_16x16x32_bf16(
      /*neg_a=*/false, a, /*neg_b=*/false, b,
      /*c_mod=*/(short)0, c, /*reuse_a=*/false, /*reuse_b=*/false);
}

// --------------------- pack: f32 [rows,cols] -> bf16 [cols,rows] -----------
__global__ void __launch_bounds__(256)
transpose_f32_to_bf16(const float* __restrict__ in, __bf16* __restrict__ out,
                      int rows, int cols, long inBatchStride, long outBatchStride) {
  __shared__ float tile[32][33];
  const float* src = in  + (long)blockIdx.z * inBatchStride;
  __bf16*      dst = out + (long)blockIdx.z * outBatchStride;
  const int c0 = blockIdx.x * 32;
  const int r0 = blockIdx.y * 32;
  for (int i = threadIdx.y; i < 32; i += 8)
    tile[i][threadIdx.x] = src[(long)(r0 + i) * cols + c0 + threadIdx.x];
  __syncthreads();
  for (int i = threadIdx.y; i < 32; i += 8)
    dst[(long)(c0 + i) * rows + r0 + threadIdx.x] = (__bf16)tile[threadIdx.x][i];
}

// ------------------------------ QKV projection -----------------------------
// Y[b, s, j] = X[b, s, :] . Wt[j, :] + bias[j]   (X = q or kv depending on j)
// grid = (24 n-blocks of 64, 16 m-blocks of 64, B); block = 128 (4 waves).
// Wave w computes the 16x64 tile at rows mblk*64 + w*16.
__global__ void __launch_bounds__(128)
qkv_gemm_kernel(const __bf16* __restrict__ qbf,    // [B][S][C]
                const __bf16* __restrict__ kvbf,   // [B][S][C]
                const __bf16* __restrict__ Wt,     // [3C][C]
                const float*  __restrict__ bias,   // [3C]
                __bf16* __restrict__ Qh,           // [B][NH][S][HD]
                __bf16* __restrict__ Kh,           // [B][NH][S][HD]
                __bf16* __restrict__ Vt)           // [B][NH][HD][S]
{
  const int lane  = threadIdx.x & 31;
  const int wave  = threadIdx.x >> 5;
  const int jbase = blockIdx.x * 64;                       // output column base
  const int m0    = blockIdx.y * 64 + wave * 16;           // output row base
  const int b     = blockIdx.z;
  const __bf16* X = (jbase < Cc ? qbf : kvbf) + (long)b * Ss * Cc;

  v8f acc[4];
  #pragma unroll
  for (int t = 0; t < 4; ++t) {
    const float bv = bias[jbase + t * 16 + (lane & 15)];   // C col = bias[n]
    #pragma unroll
    for (int r = 0; r < 8; ++r) acc[t][r] = bv;
  }

  for (int k0 = 0; k0 < Cc; k0 += 32) {
    const v16bf a = load_a_frag(X, Cc, m0, k0);
    #pragma unroll
    for (int t = 0; t < 4; ++t) {
      const v16bf bb = load_b_fragT(Wt, Cc, k0, jbase + t * 16);
      acc[t] = wmma_bf16(a, bb, acc[t]);
    }
  }

  const int half = lane >> 4;
  #pragma unroll
  for (int t = 0; t < 4; ++t) {
    const int j = jbase + t * 16 + (lane & 15);
    const int h = (j & (Cc - 1)) >> 6;                     // head within Q/K/V
    const int d = j & (HD - 1);
    #pragma unroll
    for (int r = 0; r < 8; ++r) {
      const int s = m0 + r + 8 * half;                     // D layout row
      const __bf16 v = (__bf16)acc[t][r];
      if (j < Cc)            Qh[(((long)b * NH + h) * Ss + s) * HD + d] = v;
      else if (j < 2 * Cc)   Kh[(((long)b * NH + h) * Ss + s) * HD + d] = v;
      else                   Vt[(((long)b * NH + h) * HD + d) * Ss + s] = v;
    }
  }
}

// ------------------------------ flash attention ----------------------------
// grid = (S/64, NH, B); block = 128 (4 waves). Wave owns 16 query rows.
// Per 32-key step: 4 QK^T WMMAs, online softmax, P transposed D->A through a
// wave-private LDS tile, 4 PV WMMAs into the running f32 accumulator.
__global__ void __launch_bounds__(128)
attn_kernel(const __bf16* __restrict__ Qh,   // [B][NH][S][HD]
            const __bf16* __restrict__ Kh,   // [B][NH][S][HD]
            const __bf16* __restrict__ Vt,   // [B][NH][HD][S]
            float* __restrict__ out)         // [B][C][H][W] == [B][C][S]
{
  __shared__ __align__(16) __bf16 plds_all[4][16 * 40];    // 40-elem row pitch
  const int lane = threadIdx.x & 31;
  const int wave = threadIdx.x >> 5;
  __bf16* plds = plds_all[wave];

  const int h = blockIdx.y, b = blockIdx.z;
  const long bh = (long)b * NH + h;
  const __bf16* Q  = Qh + bh * Ss * HD;
  const __bf16* K  = Kh + bh * Ss * HD;
  const __bf16* VT = Vt + bh * HD * Ss;
  const int m0   = blockIdx.x * 64 + wave * 16;
  const int half = lane >> 4;

  // Q fragments for this wave's 16 rows, held in registers all loop long.
  const v16bf aq0 = load_a_frag(Q, HD, m0, 0);
  const v16bf aq1 = load_a_frag(Q, HD, m0, 32);

  v8f acc[4];
  float mrun[8], lrun[8];
  #pragma unroll
  for (int t = 0; t < 4; ++t)
    #pragma unroll
    for (int r = 0; r < 8; ++r) acc[t][r] = 0.0f;
  #pragma unroll
  for (int r = 0; r < 8; ++r) { mrun[r] = -1e30f; lrun[r] = 0.0f; }

  for (int kb = 0; kb < Ss; kb += 32) {
    v8f s0, s1;
    #pragma unroll
    for (int r = 0; r < 8; ++r) { s0[r] = 0.0f; s1[r] = 0.0f; }
    // scores for keys [kb, kb+16) and [kb+16, kb+32); K-dim = HD in 2 chunks
    s0 = wmma_bf16(aq0, load_b_fragT(K, HD, 0,  kb),      s0);
    s0 = wmma_bf16(aq1, load_b_fragT(K, HD, 32, kb),      s0);
    s1 = wmma_bf16(aq0, load_b_fragT(K, HD, 0,  kb + 16), s1);
    s1 = wmma_bf16(aq1, load_b_fragT(K, HD, 32, kb + 16), s1);

    float p0[8], p1[8], corr[8];
    #pragma unroll
    for (int r = 0; r < 8; ++r) {
      const float a0 = s0[r] * SCALE, a1 = s1[r] * SCALE;
      float mx = fmaxf(a0, a1);                 // reduce over 32 keys:
      mx = fmaxf(mx, __shfl_xor(mx, 1));        // xor masks <16 stay inside the
      mx = fmaxf(mx, __shfl_xor(mx, 2));        // 16-lane N-group, so each row's
      mx = fmaxf(mx, __shfl_xor(mx, 4));        // stats reduce exactly.
      mx = fmaxf(mx, __shfl_xor(mx, 8));
      const float mnew = fmaxf(mrun[r], mx);
      p0[r] = __expf(a0 - mnew);
      p1[r] = __expf(a1 - mnew);
      float rs = p0[r] + p1[r];
      rs += __shfl_xor(rs, 1); rs += __shfl_xor(rs, 2);
      rs += __shfl_xor(rs, 4); rs += __shfl_xor(rs, 8);
      corr[r] = __expf(mrun[r] - mnew);
      lrun[r] = lrun[r] * corr[r] + rs;
      mrun[r] = mnew;
    }
    #pragma unroll
    for (int t = 0; t < 4; ++t)
      #pragma unroll
      for (int r = 0; r < 8; ++r) acc[t][r] *= corr[r];

    // P: D-layout (lane=col, vgpr=row) -> LDS -> A-layout (lane=row, K packed)
    {
      const int n = lane & 15;
      #pragma unroll
      for (int r = 0; r < 8; ++r) {
        const int row = r + 8 * half;
        plds[row * 40 + n]      = (__bf16)p0[r];
        plds[row * 40 + 16 + n] = (__bf16)p1[r];
      }
    }
    const v16bf ap = load_a_frag(plds, 40, 0, 0);   // same-wave LDS: ds waits
                                                    // inserted by compiler,
                                                    // DS ops in-order per wave
    #pragma unroll
    for (int t = 0; t < 4; ++t) {
      const v16bf bv = load_b_fragT(VT, Ss, kb, t * 16);
      acc[t] = wmma_bf16(ap, bv, acc[t]);
    }
  }

  // epilogue: normalize and write [B, C, S] (== [B,C,H,W]) in f32
  #pragma unroll
  for (int t = 0; t < 4; ++t) {
    const int c = h * HD + t * 16 + (lane & 15);
    #pragma unroll
    for (int r = 0; r < 8; ++r) {
      const int s = m0 + r + 8 * half;
      out[((long)b * Cc + c) * Ss + s] = acc[t][r] / lrun[r];
    }
  }
}

// --------------------------------- launch ----------------------------------
extern "C" void kernel_launch(void* const* d_in, const int* in_sizes, int n_in,
                              void* d_out, int out_size, void* d_ws, size_t ws_size,
                              hipStream_t stream) {
  const float* query     = (const float*)d_in[0];   // [B,C,H,W]
  const float* key_value = (const float*)d_in[1];   // [B,C,H,W]
  const float* W_qkv     = (const float*)d_in[2];   // [C, 3C]
  const float* b_qkv     = (const float*)d_in[3];   // [3C]
  float* out = (float*)d_out;

  // workspace carve-up (256B aligned)
  char* ws = (char*)d_ws;
  size_t off = 0;
  auto carve = [&](size_t bytes) {
    void* p = ws + off;
    off += (bytes + 255) & ~(size_t)255;
    return p;
  };
  __bf16* qbf  = (__bf16*)carve((size_t)Bb * Ss * Cc * 2);      // [B][S][C]
  __bf16* kvbf = (__bf16*)carve((size_t)Bb * Ss * Cc * 2);      // [B][S][C]
  __bf16* Wt   = (__bf16*)carve((size_t)3 * Cc * Cc * 2);       // [3C][C]
  __bf16* Qh   = (__bf16*)carve((size_t)Bb * NH * Ss * HD * 2); // [B][NH][S][HD]
  __bf16* Kh   = (__bf16*)carve((size_t)Bb * NH * Ss * HD * 2); // [B][NH][S][HD]
  __bf16* Vt   = (__bf16*)carve((size_t)Bb * NH * HD * Ss * 2); // [B][NH][HD][S]
  (void)in_sizes; (void)n_in; (void)out_size; (void)ws_size;

  // 1) pack + transpose + f32->bf16
  transpose_f32_to_bf16<<<dim3(Ss / 32, Cc / 32, Bb), dim3(32, 8), 0, stream>>>(
      query, qbf, Cc, Ss, (long)Cc * Ss, (long)Ss * Cc);
  transpose_f32_to_bf16<<<dim3(Ss / 32, Cc / 32, Bb), dim3(32, 8), 0, stream>>>(
      key_value, kvbf, Cc, Ss, (long)Cc * Ss, (long)Ss * Cc);
  transpose_f32_to_bf16<<<dim3(3 * Cc / 32, Cc / 32, 1), dim3(32, 8), 0, stream>>>(
      W_qkv, Wt, Cc, 3 * Cc, 0, 0);

  // 2) fused QKV projection (bias folded into WMMA accumulator init)
  qkv_gemm_kernel<<<dim3(3 * Cc / 64, Ss / 64, Bb), dim3(128), 0, stream>>>(
      qbf, kvbf, Wt, b_qkv, Qh, Kh, Vt);

  // 3) flash attention + output layout restore
  attn_kernel<<<dim3(Ss / 64, NH, Bb), dim3(128), 0, stream>>>(Qh, Kh, Vt, out);
}
